// GIN_Model_40458591929161
// MI455X (gfx1250) — matrix operand order repaired and checked by reading, not compiled, not run on hardware
//
#include <hip/hip_runtime.h>
#include <hip/hip_bf16.h>

typedef __attribute__((ext_vector_type(16))) _Float16 v16h;
typedef __attribute__((ext_vector_type(8)))  float    v8f;

#define WMMA_TM 128
#define WMMA_TN 64
#define WMMA_TK 32

#if __has_builtin(__builtin_amdgcn_global_load_async_to_lds_b128)
#define HAVE_ASYNC_LDS 1
typedef int int4v __attribute__((ext_vector_type(4)));
typedef __attribute__((address_space(1))) int4v g1_int4;   // global
typedef __attribute__((address_space(3))) int4v l3_int4;   // LDS
#else
#define HAVE_ASYNC_LDS 0
#endif

__device__ __forceinline__ void wait_async_then_sync()
{
#if HAVE_ASYNC_LDS
#if __has_builtin(__builtin_amdgcn_s_wait_asynccnt)
    __builtin_amdgcn_s_wait_asynccnt(0);
#else
    asm volatile("s_wait_asynccnt 0x0" ::: "memory");
#endif
#endif
    __syncthreads();
}

// ---------------------------------------------------------------------------
// Async (or vector-copy) staging of one 128x32 A tile + 64x32 B tile
// ---------------------------------------------------------------------------
__device__ __forceinline__ void stage_tiles(const _Float16* __restrict__ A, int lda,
                                            int m0,
                                            const _Float16* __restrict__ Wt, int ldw,
                                            int n0, int k0, int tid,
                                            _Float16 (&dA)[WMMA_TM][WMMA_TK],
                                            _Float16 (&dB)[WMMA_TN][WMMA_TK])
{
    // A tile: 128 rows x 64B, 2 x 16B chunks per thread
    #pragma unroll
    for (int p = 0; p < 2; ++p) {
        int i = tid + p * 256;
        int r = i >> 2, q = i & 3;
        const _Float16* ga = A + (size_t)(m0 + r) * lda + k0 + q * 8;
        _Float16* la = &dA[r][q * 8];
#if HAVE_ASYNC_LDS
        __builtin_amdgcn_global_load_async_to_lds_b128(
            (g1_int4*)ga, (l3_int4*)la, 0, 0);
#else
        *(uint4*)la = *(const uint4*)ga;
#endif
    }
    // B tile: 64 rows x 64B, 1 x 16B chunk per thread
    {
        int r = tid >> 2, q = tid & 3;
        const _Float16* gb = Wt + (size_t)(n0 + r) * ldw + k0 + q * 8;
        _Float16* lb = &dB[r][q * 8];
#if HAVE_ASYNC_LDS
        __builtin_amdgcn_global_load_async_to_lds_b128(
            (g1_int4*)gb, (l3_int4*)lb, 0, 0);
#else
        *(uint4*)lb = *(const uint4*)gb;
#endif
    }
}

// ---------------------------------------------------------------------------
// WMMA compute step (8 waves, each: 1 M-subtile x 4 N-subtiles)
// ---------------------------------------------------------------------------
__device__ __forceinline__ void wmma_step(const _Float16 (&sA)[WMMA_TM][WMMA_TK],
                                          const _Float16 (&sB)[WMMA_TN][WMMA_TK],
                                          int arow, int ak, int bcl, int bk,
                                          v8f acc[4])
{
    v16h afrag;
    {
        _Float16* ap = (_Float16*)&afrag;
        #pragma unroll
        for (int i = 0; i < 8; ++i) {
            ap[i]     = sA[arow][ak + i];        // contiguous 16B
            ap[i + 8] = sA[arow][ak + 16 + i];   // contiguous 16B
        }
    }
    #pragma unroll
    for (int nt = 0; nt < 4; ++nt) {
        v16h bfrag;
        _Float16* bp = (_Float16*)&bfrag;
        #pragma unroll
        for (int i = 0; i < 8; ++i) {
            bp[i]     = sB[nt * 16 + bcl][bk + i];        // contiguous 16B
            bp[i + 8] = sB[nt * 16 + bcl][bk + 16 + i];   // contiguous 16B
        }
        acc[nt] = __builtin_amdgcn_wmma_f32_16x16x32_f16(
            false, afrag, false, bfrag, (short)0, acc[nt], false, false);
    }
}

// ---------------------------------------------------------------------------
// f16 GEMM:  C[M,N] = act(A[M,Kp] @ Wt[N,Kp]^T + bias)
// A: f16, row stride lda (Kp, multiple of 32, zero-padded cols).
// Wt: f16 pre-transposed weights, row n holds K-dim, stride ldw (=Kp).
// Output: f32 (Cf, stride N) or f16 (Ch, stride ldc; pad cols zero-filled).
// Interior blocks: double-buffered async-to-LDS pipeline.
// ---------------------------------------------------------------------------
__launch_bounds__(256)
__global__ void gemm_f16(const _Float16* __restrict__ A, int lda,
                         const _Float16* __restrict__ Wt, int ldw,
                         const float* __restrict__ bias,
                         float* __restrict__ Cf,
                         _Float16* __restrict__ Ch, int ldc,
                         int M, int Kp, int N, int relu)
{
    __shared__ _Float16 sA[2][WMMA_TM][WMMA_TK];   // 2 x 8 KB
    __shared__ _Float16 sB[2][WMMA_TN][WMMA_TK];   // 2 x 4 KB

    const int tid  = threadIdx.x;
    const int wave = tid >> 5;
    const int lane = tid & 31;
    const int m0 = blockIdx.x * WMMA_TM;
    const int n0 = blockIdx.y * WMMA_TN;

    v8f acc[4] = {};

    const int arow = wave * 16 + (lane & 15);
    const int ak   = (lane & 16) ? 8  : 0;
    const int bk   = (lane & 16) ? 8  : 0;
    const int bcl  = lane & 15;

    const bool interior = (m0 + WMMA_TM <= M) && (n0 + WMMA_TN <= N);

    if (interior) {
        const int nk = Kp >> 5;
        stage_tiles(A, lda, m0, Wt, ldw, n0, 0, tid, sA[0], sB[0]);
        wait_async_then_sync();
        for (int t = 0; t < nk; ++t) {
            const int cur = t & 1;
            const bool more = (t + 1) < nk;
            if (more)
                stage_tiles(A, lda, m0, Wt, ldw, n0, (t + 1) * WMMA_TK, tid,
                            sA[1 - cur], sB[1 - cur]);   // overlap with compute
            wmma_step(sA[cur], sB[cur], arow, ak, bcl, bk, acc);
            if (more) wait_async_then_sync();
        }
    } else {
        // guarded path: only the row index can go OOB (Kp padded).
        // Clamp row, load unconditionally, select zero afterward.
        for (int k0 = 0; k0 < Kp; k0 += WMMA_TK) {
            {
                _Float16 ra[16];
                #pragma unroll
                for (int j = 0; j < 16; ++j) {
                    int i = tid + j * 256;
                    int gr = m0 + (i >> 5), gc = k0 + (i & 31);
                    int rc = gr < M ? gr : (M - 1);
                    _Float16 v = A[(size_t)rc * lda + gc];
                    ra[j] = (gr < M) ? v : (_Float16)0.0f;
                }
                #pragma unroll
                for (int j = 0; j < 16; ++j) {
                    int i = tid + j * 256;
                    sA[0][i >> 5][i & 31] = ra[j];
                }
            }
            {
                _Float16 rb[8];
                #pragma unroll
                for (int j = 0; j < 8; ++j) {
                    int i = tid + j * 256;
                    int gn = n0 + (i >> 5), gc = k0 + (i & 31);
                    int nc = gn < N ? gn : (N - 1);
                    _Float16 v = Wt[(size_t)nc * ldw + gc];
                    rb[j] = (gn < N) ? v : (_Float16)0.0f;
                }
                #pragma unroll
                for (int j = 0; j < 8; ++j) {
                    int i = tid + j * 256;
                    sB[0][i >> 5][i & 31] = rb[j];
                }
            }
            __syncthreads();
            wmma_step(sA[0], sB[0], arow, ak, bcl, bk, acc);
            __syncthreads();
        }
    }

    // C/D layout: elem v: lane<16 -> (M=v, N=lane); lane>=16 -> (M=v+8, N=lane-16)
    const int rb2 = m0 + wave * 16 + ((lane & 16) ? 8 : 0);
    #pragma unroll
    for (int nt = 0; nt < 4; ++nt) {
        int gn = n0 + nt * 16 + bcl;
        if (Cf) {
            if (gn < N) {
                float bb = bias[gn];
                #pragma unroll
                for (int v = 0; v < 8; ++v) {
                    int gm = rb2 + v;
                    if (gm < M) {
                        float val = acc[nt][v] + bb;
                        if (relu) val = fmaxf(val, 0.0f);
                        Cf[(size_t)gm * N + gn] = val;
                    }
                }
            }
        } else {
            if (gn < ldc) {                       // zero-fill pad columns
                bool in = gn < N;
                float bb = in ? bias[gn] : 0.0f;
                #pragma unroll
                for (int v = 0; v < 8; ++v) {
                    int gm = rb2 + v;
                    if (gm < M) {
                        float val = acc[nt][v] + bb;
                        if (relu) val = fmaxf(val, 0.0f);
                        Ch[(size_t)gm * ldc + gn] = in ? (_Float16)val
                                                       : (_Float16)0.0f;
                    }
                }
            }
        }
    }
}

// ---------------------------------------------------------------------------
// Elementwise / scatter / reduction helpers
// ---------------------------------------------------------------------------
__global__ void k_copy(const float* __restrict__ s, float* __restrict__ d, size_t n)
{
    size_t i = blockIdx.x * (size_t)blockDim.x + threadIdx.x;
    size_t st = (size_t)gridDim.x * blockDim.x;
    for (; i < n; i += st) d[i] = s[i];
}

__global__ void k_zero(float* __restrict__ d, size_t n)
{
    size_t i = blockIdx.x * (size_t)blockDim.x + threadIdx.x;
    size_t st = (size_t)gridDim.x * blockDim.x;
    for (; i < n; i += st) d[i] = 0.0f;
}

// f32 [rows,K] -> f16 [rows,Kp], pad cols zeroed
__global__ void k_cvt_f16_pad(const float* __restrict__ s, _Float16* __restrict__ d,
                              int rows, int K, int Kp)
{
    size_t total = (size_t)rows * Kp;
    size_t i = blockIdx.x * (size_t)blockDim.x + threadIdx.x;
    size_t st = (size_t)gridDim.x * blockDim.x;
    for (; i < total; i += st) {
        size_t r = i / (size_t)Kp;
        int    c = (int)(i - r * (size_t)Kp);
        d[i] = (c < K) ? (_Float16)s[r * (size_t)K + c] : (_Float16)0.0f;
    }
}

// W f32 [K,N] -> Wt f16 [N,Kp] (transposed, pad cols zeroed)
__global__ void k_cvt_w_t(const float* __restrict__ w, _Float16* __restrict__ wt,
                          int K, int N, int Kp)
{
    size_t total = (size_t)N * Kp;
    size_t i = blockIdx.x * (size_t)blockDim.x + threadIdx.x;
    size_t st = (size_t)gridDim.x * blockDim.x;
    for (; i < total; i += st) {
        size_t n = i / (size_t)Kp;
        int    k = (int)(i - n * (size_t)Kp);
        wt[i] = (k < K) ? (_Float16)w[(size_t)k * N + n] : (_Float16)0.0f;
    }
}

// h[dst[e]][c] += x[src[e]][c]
__global__ void k_scatter_add(const float* __restrict__ x,
                              const int* __restrict__ se,
                              const int* __restrict__ de,
                              float* __restrict__ h, int E, int d)
{
    size_t total = (size_t)E * d;
    size_t i = blockIdx.x * (size_t)blockDim.x + threadIdx.x;
    size_t st = (size_t)gridDim.x * blockDim.x;
    for (; i < total; i += st) {
        size_t e = i / (size_t)d;
        int    c = (int)(i - e * (size_t)d);
        atomicAdd(&h[(size_t)de[e] * d + c], x[(size_t)se[e] * d + c]);
    }
}

__launch_bounds__(256)
__global__ void k_bn_partial(const float* __restrict__ x, int rows, int C,
                             float* __restrict__ sums /* [2*C] zeroed */)
{
    __shared__ float ls[512], ls2[512];
    for (int c = threadIdx.x; c < C; c += 256) { ls[c] = 0.0f; ls2[c] = 0.0f; }
    __syncthreads();

    size_t total = (size_t)rows * C;
    size_t i = blockIdx.x * (size_t)blockDim.x + threadIdx.x;
    size_t st = (size_t)gridDim.x * blockDim.x;
    for (; i < total; i += st) {
        float v = x[i];                       // coalesced
        int c = (int)(i % (size_t)C);
        atomicAdd(&ls[c], v);                 // ds_add_f32
        atomicAdd(&ls2[c], v * v);
    }
    __syncthreads();
    for (int c = threadIdx.x; c < C; c += 256) {
        atomicAdd(&sums[c], ls[c]);
        atomicAdd(&sums[C + c], ls2[c]);
    }
}

__global__ void k_bn_finalize(const float* __restrict__ sums, int rows, int C,
                              float* __restrict__ mu, float* __restrict__ iv)
{
    int c = blockIdx.x * blockDim.x + threadIdx.x;
    if (c < C) {
        float m   = sums[c] / (float)rows;
        float var = sums[C + c] / (float)rows - m * m;
        mu[c] = m;
        iv[c] = rsqrtf(var + 1e-5f);
    }
}

__global__ void k_bn_apply(float* __restrict__ x, int rows, int C,
                           const float* __restrict__ mu, const float* __restrict__ iv,
                           const float* __restrict__ g, const float* __restrict__ b)
{
    size_t total = (size_t)rows * C;
    size_t i = blockIdx.x * (size_t)blockDim.x + threadIdx.x;
    size_t st = (size_t)gridDim.x * blockDim.x;
    for (; i < total; i += st) {
        int c = (int)(i % (size_t)C);
        x[i] = (x[i] - mu[c]) * iv[c] * g[c] + b[c];
    }
}

// g[batch[r]][c] += x[r][c]
__global__ void k_pool(const float* __restrict__ x, const int* __restrict__ batch,
                       float* __restrict__ g, int rows, int C)
{
    size_t total = (size_t)rows * C;
    size_t i = blockIdx.x * (size_t)blockDim.x + threadIdx.x;
    size_t st = (size_t)gridDim.x * blockDim.x;
    for (; i < total; i += st) {
        size_t r = i / (size_t)C;
        int    c = (int)(i - r * (size_t)C);
        atomicAdd(&g[(size_t)batch[r] * C + c], x[i]);
    }
}

// cat f16 [G,2272]: cols 0..1023 = g1 (f16, ld 1024); 1024..2240 = ecfp; rest 0
__global__ void k_concat_f16(const _Float16* __restrict__ g1,
                             const float* __restrict__ ecfp,
                             _Float16* __restrict__ o, int G)
{
    const int CP = 2272, CA = 1024, CE = 1217;
    size_t total = (size_t)G * CP;
    size_t i = blockIdx.x * (size_t)blockDim.x + threadIdx.x;
    size_t st = (size_t)gridDim.x * blockDim.x;
    for (; i < total; i += st) {
        size_t r = i / (size_t)CP;
        int    c = (int)(i - r * (size_t)CP);
        _Float16 v;
        if (c < CA)            v = g1[r * (size_t)CA + c];
        else if (c < CA + CE)  v = (_Float16)ecfp[r * (size_t)CE + (c - CA)];
        else                   v = (_Float16)0.0f;
        o[i] = v;
    }
}

// ---------------------------------------------------------------------------
// Host-side orchestration
// ---------------------------------------------------------------------------
static inline int eltgrid(size_t n)
{
    size_t b = (n + 255) / 256;
    if (b > (1u << 20)) b = (1u << 20);
    if (b == 0) b = 1;
    return (int)b;
}

static inline int ru32(int x) { return (x + 31) & ~31; }

static inline void launch_gemm(hipStream_t s, const _Float16* A, int lda,
                               const _Float16* Wt, int ldw, const float* bias,
                               float* Cf, _Float16* Ch, int ldc,
                               int M, int Kp, int N, int relu)
{
    dim3 grid((M + WMMA_TM - 1) / WMMA_TM, (N + WMMA_TN - 1) / WMMA_TN);
    gemm_f16<<<grid, 256, 0, s>>>(A, lda, Wt, ldw, bias, Cf, Ch, ldc,
                                  M, Kp, N, relu);
}

extern "C" void kernel_launch(void* const* d_in, const int* in_sizes, int n_in,
                              void* d_out, int out_size, void* d_ws, size_t ws_size,
                              hipStream_t stream)
{
    (void)in_sizes; (void)n_in; (void)out_size; (void)ws_size;

    const int N_NODES = 50000, N_EDGES = 800000, G = 1024;
    const int din[5] = { 57, 114, 228, 256, 256 };
    const int dh [5] = { 114, 228, 256, 256, 512 };

    // Input layout: recursive dict insertion order.
    const float* x_in = (const float*)d_in[0];
    const float* ecfp = (const float*)d_in[1];
    const float* ginW = (const float*)d_in[32];
    const float* ginB = (const float*)d_in[33];
    const float* f1W  = (const float*)d_in[34];
    const float* f1b  = (const float*)d_in[35];
    const float* f2W  = (const float*)d_in[36];
    const float* f2b  = (const float*)d_in[37];
    const float* f3W  = (const float*)d_in[38];
    const float* f3b  = (const float*)d_in[39];
    const float* f4W  = (const float*)d_in[40];
    const float* f4b  = (const float*)d_in[41];
    const float* f5W  = (const float*)d_in[42];
    const float* f5b  = (const float*)d_in[43];
    const int* srcE   = (const int*)d_in[44];
    const int* dstE   = (const int*)d_in[45];
    const int* batch  = (const int*)d_in[46];

    const size_t NODE_F32 = (size_t)N_NODES * 512;   // floats
    const size_t NODE_F16 = (size_t)N_NODES * 512;   // halves
    const size_t WT_HALVES = 2400000;                // >= 1024*2272

    float*    bufA = (float*)d_ws;                   // x / t2 / BN (f32)
    float*    bufB = bufA + NODE_F32;                // h f32 / pooled g f32
    _Float16* Ah   = (_Float16*)(bufB + NODE_F32);   // f16 GEMM input A
    _Float16* Th   = Ah + NODE_F16;                  // f16 chained outputs
    _Float16* Wt   = Th + NODE_F16;                  // f16 transposed weights
    float*    mu   = (float*)(Wt + WT_HALVES);
    float*    iv   = mu + 512;
    float*    sums = iv + 512;

    // ---- 5 GINConv layers ----
    for (int l = 0; l < 5; ++l) {
        const float* xcur = (l == 0) ? x_in : bufA;
        int d = din[l], h = dh[l];
        int dp = ru32(d), hp = ru32(h);

        const float* W1 = (const float*)d_in[2 + 6 * l + 0];
        const float* b1 = (const float*)d_in[2 + 6 * l + 1];
        const float* W2 = (const float*)d_in[2 + 6 * l + 2];
        const float* b2 = (const float*)d_in[2 + 6 * l + 3];
        const float* gm = (const float*)d_in[2 + 6 * l + 4];
        const float* bt = (const float*)d_in[2 + 6 * l + 5];

        size_t nd = (size_t)N_NODES * d;
        size_t ed = (size_t)N_EDGES * d;

        // h = x + segment_sum(x[src] -> dst)   (f32 accumulation)
        k_copy<<<eltgrid(nd), 256, 0, stream>>>(xcur, bufB, nd);
        k_scatter_add<<<eltgrid(ed), 256, 0, stream>>>(xcur, srcE, dstE, bufB,
                                                       N_EDGES, d);
        // convert h -> f16 padded; t1 = relu(h @ W1 + b1) stored f16 padded
        k_cvt_f16_pad<<<eltgrid((size_t)N_NODES * dp), 256, 0, stream>>>(
            bufB, Ah, N_NODES, d, dp);
        k_cvt_w_t<<<eltgrid((size_t)h * dp), 256, 0, stream>>>(W1, Wt, d, h, dp);
        launch_gemm(stream, Ah, dp, Wt, dp, b1, nullptr, Th, hp,
                    N_NODES, dp, h, 1);
        // t2 = t1 @ W2 + b2  (f32, overwrites x slot)
        k_cvt_w_t<<<eltgrid((size_t)h * hp), 256, 0, stream>>>(W2, Wt, h, h, hp);
        launch_gemm(stream, Th, hp, Wt, hp, b2, bufA, nullptr, 0,
                    N_NODES, hp, h, 0);
        // BatchNorm (batch stats), in place
        k_zero<<<eltgrid(2 * h), 256, 0, stream>>>(sums, (size_t)(2 * h));
        k_bn_partial<<<512, 256, 0, stream>>>(bufA, N_NODES, h, sums);
        k_bn_finalize<<<(h + 255) / 256, 256, 0, stream>>>(sums, N_NODES, h, mu, iv);
        k_bn_apply<<<eltgrid((size_t)N_NODES * h), 256, 0, stream>>>(
            bufA, N_NODES, h, mu, iv, gm, bt);
    }

    // ---- global_add_pool -> g[1024,512] f32 in bufB ----
    k_zero<<<eltgrid((size_t)G * 512), 256, 0, stream>>>(bufB, (size_t)G * 512);
    k_pool<<<eltgrid((size_t)N_NODES * 512), 256, 0, stream>>>(bufA, batch, bufB,
                                                               N_NODES, 512);

    // ---- head MLP (f16 chain, f32 final) ----
    k_cvt_f16_pad<<<eltgrid((size_t)G * 512), 256, 0, stream>>>(bufB, Ah, G, 512, 512);
    k_cvt_w_t<<<eltgrid((size_t)1024 * 512), 256, 0, stream>>>(ginW, Wt, 512, 1024, 512);
    launch_gemm(stream, Ah, 512, Wt, 512, ginB, nullptr, Th, 1024, G, 512, 1024, 1);

    k_concat_f16<<<eltgrid((size_t)G * 2272), 256, 0, stream>>>(Th, ecfp, Ah, G);

    k_cvt_w_t<<<eltgrid((size_t)1024 * 2272), 256, 0, stream>>>(f1W, Wt, 2241, 1024, 2272);
    launch_gemm(stream, Ah, 2272, Wt, 2272, f1b, nullptr, Th, 1024, G, 2272, 1024, 1);

    k_cvt_w_t<<<eltgrid((size_t)2048 * 1024), 256, 0, stream>>>(f2W, Wt, 1024, 2048, 1024);
    launch_gemm(stream, Th, 1024, Wt, 1024, f2b, nullptr, Ah, 2048, G, 1024, 2048, 1);

    k_cvt_w_t<<<eltgrid((size_t)1024 * 2048), 256, 0, stream>>>(f3W, Wt, 2048, 1024, 2048);
    launch_gemm(stream, Ah, 2048, Wt, 2048, f3b, nullptr, Th, 1024, G, 2048, 1024, 1);

    k_cvt_w_t<<<eltgrid((size_t)256 * 1024), 256, 0, stream>>>(f4W, Wt, 1024, 256, 1024);
    launch_gemm(stream, Th, 1024, Wt, 1024, f4b, nullptr, Ah, 256, G, 1024, 256, 1);

    k_cvt_w_t<<<eltgrid((size_t)1 * 256), 256, 0, stream>>>(f5W, Wt, 256, 1, 256);
    launch_gemm(stream, Ah, 256, Wt, 256, f5b, (float*)d_out, nullptr, 0,
                G, 256, 1, 0);
}
